// GNN_node_19198503813152
// MI455X (gfx1250) — matrix operand order repaired and checked by reading, not compiled, not run on hardware
//
#include <hip/hip_runtime.h>
#include <hip/hip_bf16.h>

// GIN (5-layer) forward for MI455X / gfx1250, wave32, bf16 WMMA GEMMs.
//
// Per layer:
//   zero(agg,stats) -> bond+msg+scatter (f32 atomics; h is L2-resident) ->
//   GEMM1 [fused A-gen: bf16((1+eps)*h+agg)] -> t1, fused column stats ->
//   bnfinal1 -> GEMM2 [fused A-gen: bf16(relu(a1*t1+c1))] -> t2, fused stats ->
//   bnfinal2 -> apply outer BN(+relu) -> h (d_out on last layer)
//
// All row-dim buffers padded to NPAD=150016 (=1172*128) so WMMA tile loads and
// stores are unconditional. Pad rows are excluded from the fused BN statistics
// with hoisted branch-free multiply masks.
// Linear biases b1/b2 are skipped: a uniform per-column shift cancels in BN.

#define NN   150000
#define NPAD 150016          // 1172 * 128
#define EE   600000
#define DD   128

typedef __attribute__((ext_vector_type(16))) __bf16 v16bf;
typedef __attribute__((ext_vector_type(8)))  __bf16 v8bf;
typedef __attribute__((ext_vector_type(8)))  float  v8f;

// pack two f32 -> packed bf16 pair (round-to-nearest-even), pure integer ops
static __device__ __forceinline__ unsigned pkbf(float a, float b) {
  unsigned ua = __builtin_bit_cast(unsigned, a);
  unsigned ub = __builtin_bit_cast(unsigned, b);
  ua = (ua + 0x7FFFu + ((ua >> 16) & 1u)) >> 16;
  ub = (ub + 0x7FFFu + ((ub >> 16) & 1u)) & 0xFFFF0000u;
  return ua | ub;
}

static __device__ __forceinline__ void pack0(float4 x, float4 y, float e, unsigned* u) {
  u[0] = pkbf(fmaf(e, x.x, y.x), fmaf(e, x.y, y.y));
  u[1] = pkbf(fmaf(e, x.z, y.z), fmaf(e, x.w, y.w));
}
static __device__ __forceinline__ void pack1(float4 x, float4 s, float4 c, unsigned* u) {
  u[0] = pkbf(fmaxf(fmaf(s.x, x.x, c.x), 0.f), fmaxf(fmaf(s.y, x.y, c.y), 0.f));
  u[1] = pkbf(fmaxf(fmaf(s.z, x.z, c.z), 0.f), fmaxf(fmaf(s.w, x.w, c.w), 0.f));
}

// ---------------------------------------------------------------- zero
__global__ void k_zero(float* __restrict__ p, long n) {
  long i = (long)blockIdx.x * blockDim.x + threadIdx.x;
  long stride = (long)gridDim.x * blockDim.x;
  for (; i < n; i += stride) p[i] = 0.0f;
}

// ---------------------------------------------------------------- atom encoder
__global__ void k_atom(const int* __restrict__ x, const float* __restrict__ aemb,
                       float* __restrict__ h) {
  long idx = (long)blockIdx.x * blockDim.x + threadIdx.x;
  int n = (int)(idx >> 5);
  int q = (int)(idx & 31);
  if (n >= NN) return;
  float s0 = 0.f, s1 = 0.f, s2 = 0.f, s3 = 0.f;
#pragma unroll
  for (int f = 0; f < 9; ++f) {
    int v = x[n * 9 + f];
    const float4 t = *(const float4*)(aemb + ((size_t)f * 120 + v) * DD + q * 4);
    s0 += t.x; s1 += t.y; s2 += t.z; s3 += t.w;
  }
  *(float4*)(h + (size_t)n * DD + q * 4) = make_float4(s0, s1, s2, s3);
}

// ---------------------------------------------------------------- edge scatter
// msg = relu(h[row] + sum_f bond_emb[f, ea[e,f]]); atomicAdd into agg[col]
__global__ void k_bond_scatter(const int* __restrict__ ei, const int* __restrict__ ea,
                               const float* __restrict__ bemb,  // layer base [3,6,128]
                               const float* __restrict__ h, float* __restrict__ agg) {
  long idx = (long)blockIdx.x * blockDim.x + threadIdx.x;
  int e = (int)(idx >> 5);
  int q = (int)(idx & 31);
  if (e >= EE) return;
  int row = ei[e];
  int col = ei[EE + e];
  int a0 = ea[e * 3 + 0], a1 = ea[e * 3 + 1], a2 = ea[e * 3 + 2];
  const float4 hm = *(const float4*)(h + (size_t)row * DD + q * 4);
  const float4 e0 = *(const float4*)(bemb + ((size_t)(0 * 6 + a0)) * DD + q * 4);
  const float4 e1 = *(const float4*)(bemb + ((size_t)(1 * 6 + a1)) * DD + q * 4);
  const float4 e2 = *(const float4*)(bemb + ((size_t)(2 * 6 + a2)) * DD + q * 4);
  float* ag = agg + (size_t)col * DD + q * 4;
  atomicAdd(ag + 0, fmaxf(hm.x + e0.x + e1.x + e2.x, 0.f));
  atomicAdd(ag + 1, fmaxf(hm.y + e0.y + e1.y + e2.y, 0.f));
  atomicAdd(ag + 2, fmaxf(hm.z + e0.z + e1.z + e2.z, 0.f));
  atomicAdd(ag + 3, fmaxf(hm.w + e0.w + e1.w + e2.w, 0.f));
}

// ---------------------------------------------------------------- W [K,NC] f32 -> W^T [NC,K] bf16
__global__ void k_wconvt(const float* __restrict__ W, __bf16* __restrict__ Wt,
                         int K, int NC) {
  int idx = blockIdx.x * blockDim.x + threadIdx.x;
  if (idx >= K * NC) return;
  int k = idx / NC, ncol = idx % NC;
  unsigned u = __builtin_bit_cast(unsigned, W[idx]);
  unsigned short hv = (unsigned short)((u + 0x7FFFu + ((u >> 16) & 1u)) >> 16);
  Wt[(size_t)ncol * K + k] = __builtin_bit_cast(__bf16, hv);
}

// ---------------------------------------------------------------- fused WMMA GEMM
// out[NPAD,NC] = A[NPAD,K] x W[K,NC]  (bf16 mul, f32 acc), A generated on the fly:
//   MODE 0: A = (1+eps[l])*X + Y          (X=h, Y=agg, Zp=eps vec)
//   MODE 1: A = relu(Y[k]*X + Zp[k])      (X=t1, Y=a1, Zp=c1; BN1+relu fused)
// Column sums/sumsqs for BatchNorm accumulate in registers during the tile
// loop, flush through LDS after it, then one global atomic per column/block.
template <int K, int NC, int MODE>
__global__ __launch_bounds__(256) void k_gemm_fused(
    const float* __restrict__ X, const float* __restrict__ Y,
    const float* __restrict__ Zp, int l,
    const __bf16* __restrict__ Bt, float* __restrict__ out,
    float* __restrict__ gsum, float* __restrict__ gsq, int nrows) {
  __shared__ float lsum[NC];
  __shared__ float lsq[NC];
  if (threadIdx.x < NC) { lsum[threadIdx.x] = 0.f; lsq[threadIdx.x] = 0.f; }
  __syncthreads();

  const int wave = threadIdx.x >> 5;
  const int lane = threadIdx.x & 31;
  const int l16  = lane & 15;
  const int half = lane >> 4;
  const int mbase = blockIdx.x * 128 + wave * 16;
  const int arow  = mbase + l16;          // buffers row-padded: no clamp needed

  constexpr int KSTEPS = K / 32;
  constexpr int NTILES = NC / 16;
  // A-operand layout (16-bit 16x32): lane holds row l16; K-chunks
  // [kk*32+half*8, +8) in elems 0..7 and [kk*32+16+half*8, +8) in elems 8..15.
  v16bf a[KSTEPS];
  {
    const float* xr = X + (size_t)arow * K;
    float e = 0.f;
    const float* yr = nullptr;
    if constexpr (MODE == 0) { e = 1.0f + Zp[l]; yr = Y + (size_t)arow * K; }
#pragma unroll
    for (int kk = 0; kk < KSTEPS; ++kk) {
      const int o0 = kk * 32 + half * 8, o1 = o0 + 16;
      float4 xa = *(const float4*)(xr + o0); float4 xb = *(const float4*)(xr + o0 + 4);
      float4 xc = *(const float4*)(xr + o1); float4 xd = *(const float4*)(xr + o1 + 4);
      union { v16bf v; unsigned u[8]; } t;
      if constexpr (MODE == 0) {
        float4 ya = *(const float4*)(yr + o0); float4 yb = *(const float4*)(yr + o0 + 4);
        float4 yc = *(const float4*)(yr + o1); float4 yd = *(const float4*)(yr + o1 + 4);
        pack0(xa, ya, e, &t.u[0]); pack0(xb, yb, e, &t.u[2]);
        pack0(xc, yc, e, &t.u[4]); pack0(xd, yd, e, &t.u[6]);
      } else {
        float4 sa = *(const float4*)(Y + o0);  float4 sb = *(const float4*)(Y + o0 + 4);
        float4 sc = *(const float4*)(Y + o1);  float4 sd = *(const float4*)(Y + o1 + 4);
        float4 ca = *(const float4*)(Zp + o0); float4 cb = *(const float4*)(Zp + o0 + 4);
        float4 cc = *(const float4*)(Zp + o1); float4 cd = *(const float4*)(Zp + o1 + 4);
        pack1(xa, sa, ca, &t.u[0]); pack1(xb, sb, cb, &t.u[2]);
        pack1(xc, sc, cc, &t.u[4]); pack1(xd, sd, cd, &t.u[6]);
      }
      a[kk] = t.v;
    }
  }

  // hoisted branch-free pad mask (only last block has vrem < 8)
  const int vrem = nrows - (mbase + half * 8);
  float mk[8];
#pragma unroll
  for (int r = 0; r < 8; ++r) mk[r] = (r < vrem) ? 1.f : 0.f;

  float sreg[NTILES], qreg[NTILES];

#pragma unroll
  for (int nt = 0; nt < NTILES; ++nt) {
    v8f acc = {};
    // B-operand layout (16-bit 32x16): lane holds col l16, K-chunks
    // [kk*32+half*8, +8) and [kk*32+16+half*8, +8) from W^T -> two b128 loads.
    const __bf16* bp = Bt + (size_t)(nt * 16 + l16) * K + half * 8;
#pragma unroll
    for (int kk = 0; kk < KSTEPS; ++kk) {
      union { v16bf v; v8bf h[2]; } b;
      b.h[0] = *(const v8bf*)(bp + kk * 32);
      b.h[1] = *(const v8bf*)(bp + kk * 32 + 16);
      acc = __builtin_amdgcn_wmma_f32_16x16x32_bf16(
          false, a[kk], false, b.v, (short)0, acc, false, false);
    }
    float* orow = out + (size_t)(mbase + half * 8) * NC + nt * 16 + l16;
    float s = 0.f, q = 0.f;
#pragma unroll
    for (int r = 0; r < 8; ++r) {
      float v = acc[r];
      orow[(size_t)r * NC] = v;            // unconditional: padded buffer
      v *= mk[r];                          // hoisted pad mask
      s += v;
      q = fmaf(v, v, q);
    }
    sreg[nt] = s;
    qreg[nt] = q;
  }

  // flush stats: LDS atomics once per wave, after the WMMA loop
#pragma unroll
  for (int nt = 0; nt < NTILES; ++nt) {
    atomicAdd(&lsum[nt * 16 + l16], sreg[nt]);
    atomicAdd(&lsq[nt * 16 + l16],  qreg[nt]);
  }

  __syncthreads();
  if (threadIdx.x < NC) {
    atomicAdd(&gsum[threadIdx.x], lsum[threadIdx.x]);
    atomicAdd(&gsq[threadIdx.x],  lsq[threadIdx.x]);
  }
}

// ---------------------------------------------------------------- BN finalize
__global__ void k_bnfinal(const float* __restrict__ sum, const float* __restrict__ sq,
                          const float* __restrict__ g, const float* __restrict__ b,
                          float* __restrict__ a, float* __restrict__ c,
                          int C, float invN) {
  int i = blockIdx.x * blockDim.x + threadIdx.x;
  if (i >= C) return;
  float mu = sum[i] * invN;
  float var = sq[i] * invN - mu * mu;
  float s = g[i] * rsqrtf(var + 1e-5f);
  a[i] = s;
  c[i] = b[i] - mu * s;
}

// outer BN (+relu except last layer) -> f32 h   (float4-vectorized)
__global__ void k_apply2(const float* __restrict__ X, const float* __restrict__ a,
                         const float* __restrict__ c, float* __restrict__ Y,
                         long n4, int relu) {
  long i = (long)blockIdx.x * blockDim.x + threadIdx.x;
  if (i >= n4) return;
  int col = (int)(i & (DD / 4 - 1)) * 4;
  float4 x  = *(const float4*)(X + i * 4);
  float4 av = *(const float4*)(a + col);
  float4 cv = *(const float4*)(c + col);
  float4 y;
  y.x = fmaf(x.x, av.x, cv.x); y.y = fmaf(x.y, av.y, cv.y);
  y.z = fmaf(x.z, av.z, cv.z); y.w = fmaf(x.w, av.w, cv.w);
  if (relu) {
    y.x = fmaxf(y.x, 0.f); y.y = fmaxf(y.y, 0.f);
    y.z = fmaxf(y.z, 0.f); y.w = fmaxf(y.w, 0.f);
  }
  *(float4*)(Y + i * 4) = y;
}

// ---------------------------------------------------------------- launch
extern "C" void kernel_launch(void* const* d_in, const int* in_sizes, int n_in,
                              void* d_out, int out_size, void* d_ws, size_t ws_size,
                              hipStream_t stream) {
  const int*   x        = (const int*)d_in[0];
  const int*   ei       = (const int*)d_in[1];
  const int*   ea       = (const int*)d_in[2];
  const float* atom_emb = (const float*)d_in[3];
  const float* bond_emb = (const float*)d_in[4];
  const float* W1       = (const float*)d_in[5];
  // d_in[6] = b1 : cancels in BN1
  const float* bn1_g    = (const float*)d_in[7];
  const float* bn1_b    = (const float*)d_in[8];
  const float* W2       = (const float*)d_in[9];
  // d_in[10] = b2 : cancels in outer BN
  const float* epsv     = (const float*)d_in[11];
  const float* bn_g     = (const float*)d_in[12];
  const float* bn_b     = (const float*)d_in[13];

  // workspace (row-padded to NPAD):
  char* ws = (char*)d_ws;
  float*  h    = (float*)(ws);                       // [NPAD,128] f32
  float*  agg  = (float*)(ws + 76808192);            // [NPAD,128] f32 (reused as t2)
  float*  t2   = agg;
  float*  t1   = (float*)(ws + 153616384);           // [NPAD,256] f32
  __bf16* wb1t = (__bf16*)(ws + 307232768);          // W1^T bf16 [256,128]
  __bf16* wb2t = (__bf16*)(ws + 307298304);          // W2^T bf16 [128,256]
  float*  st   = (float*)(ws + 307363840);
  float *sum1 = st,        *sq1 = st + 256;
  float *sum2 = st + 512,  *sq2 = st + 640;
  float *a1   = st + 768,  *c1  = st + 1024;
  float *a2   = st + 1280, *c2  = st + 1408;

  const long nPD = (long)NPAD * DD;        // padded agg size
  const long n4  = (long)NN * DD / 4;      // float4 count for apply2
  const int  gemmGrid = NPAD / 128;        // 1172

  k_atom<<<(NN * 32 + 255) / 256, 256, 0, stream>>>(x, atom_emb, h);

  for (int l = 0; l < 5; ++l) {
    k_zero<<<4096, 256, 0, stream>>>(agg, nPD);
    k_zero<<<1, 256, 0, stream>>>(st, 768);

    k_bond_scatter<<<(EE * 32 + 255) / 256, 256, 0, stream>>>(
        ei, ea, bond_emb + (size_t)l * 3 * 6 * DD, h, agg);

    k_wconvt<<<(128 * 256 + 255) / 256, 256, 0, stream>>>(W1 + (size_t)l * 128 * 256, wb1t, 128, 256);
    k_wconvt<<<(256 * 128 + 255) / 256, 256, 0, stream>>>(W2 + (size_t)l * 256 * 128, wb2t, 256, 128);

    // GEMM1: A = (1+eps)*h + agg, fused BN1 stats
    k_gemm_fused<128, 256, 0><<<gemmGrid, 256, 0, stream>>>(
        h, agg, epsv, l, wb1t, t1, sum1, sq1, NN);
    k_bnfinal<<<1, 256, 0, stream>>>(sum1, sq1, bn1_g + l * 256, bn1_b + l * 256,
                                     a1, c1, 256, 1.0f / NN);

    // GEMM2: A = relu(a1*t1 + c1), fused outer-BN stats
    k_gemm_fused<256, 128, 1><<<gemmGrid, 256, 0, stream>>>(
        t1, a1, c1, 0, wb2t, t2, sum2, sq2, NN);
    k_bnfinal<<<1, 128, 0, stream>>>(sum2, sq2, bn_g + l * 128, bn_b + l * 128,
                                     a2, c2, 128, 1.0f / NN);

    float* dst = (l == 4) ? (float*)d_out : h;
    k_apply2<<<(int)((n4 + 255) / 256), 256, 0, stream>>>(t2, a2, c2, dst, n4, l < 4);
  }
}
